// Dec_11536282157274
// MI455X (gfx1250) — matrix operand order repaired and checked by reading, not compiled, not run on hardware
//
#include <hip/hip_runtime.h>
#include <hip/hip_bf16.h>
#include <stdint.h>

// ---------- problem constants (match reference) ----------
#define ROI   14
#define PWIN  7
#define C_CH  512
#define HWDIM 64
#define R_ROI 2048
#define DDIM  4096
#define NCLS  21

typedef __attribute__((ext_vector_type(16))) __bf16 v16bf;
typedef __attribute__((ext_vector_type(8)))  float  v8f;

struct AB32 { uint4 lo, hi; };   // 32 bytes = 16 bf16 = one WMMA operand fragment per lane

static __device__ __forceinline__ unsigned short f2bf_bits(float f) {
    unsigned u = __builtin_bit_cast(unsigned, f);
    u += 0x7FFFu + ((u >> 16) & 1u);      // round-to-nearest-even
    return (unsigned short)(u >> 16);
}

// ---------------------------------------------------------------------------
// 1) fp32 -> bf16 conversion (8 elements / thread, b128 in / b128 out)
// ---------------------------------------------------------------------------
__global__ __launch_bounds__(256) void k_f32_to_bf16(const float* __restrict__ src,
                                                     unsigned short* __restrict__ dst,
                                                     int n8) {
    int i = blockIdx.x * blockDim.x + threadIdx.x;
    if (i >= n8) return;
    const float4* s = reinterpret_cast<const float4*>(src) + (size_t)i * 2;
    float4 a = s[0], b = s[1];
    uint4 o;
    o.x = (unsigned)f2bf_bits(a.x) | ((unsigned)f2bf_bits(a.y) << 16);
    o.y = (unsigned)f2bf_bits(a.z) | ((unsigned)f2bf_bits(a.w) << 16);
    o.z = (unsigned)f2bf_bits(b.x) | ((unsigned)f2bf_bits(b.y) << 16);
    o.w = (unsigned)f2bf_bits(b.z) | ((unsigned)f2bf_bits(b.w) << 16);
    reinterpret_cast<uint4*>(dst)[i] = o;
}

// ---------------------------------------------------------------------------
// 2) SPP: per (roi, channel, quadrant) 7x7 max-pool -> bf16 features [R, 4096]
// ---------------------------------------------------------------------------
__global__ __launch_bounds__(256) void k_spp(const float* __restrict__ x,
                                             const int* __restrict__ ssw,
                                             unsigned short* __restrict__ yb) {
    int r = blockIdx.x;
    int c = blockIdx.y * 64 + (threadIdx.x >> 2);
    int q = threadIdx.x & 3;
    int y0 = ssw[r * 4 + 0];
    int x0 = ssw[r * 4 + 1];
    int qy = q >> 1, qx = q & 1;
    const float* p = x + (size_t)c * (HWDIM * HWDIM)
                       + (size_t)(y0 + qy * PWIN) * HWDIM + (x0 + qx * PWIN);
    float m = -1e30f;
#pragma unroll
    for (int dy = 0; dy < PWIN; ++dy)
#pragma unroll
        for (int dx = 0; dx < PWIN; ++dx)
            m = fmaxf(m, p[dy * HWDIM + dx]);
    unsigned short bv = f2bf_bits(m);
    int f = c * 4 + q;
    yb[(size_t)r * DDIM + f]        = bv;
    yb[(size_t)r * DDIM + 2048 + f] = bv;
}

// ---------------------------------------------------------------------------
// 3) WMMA GEMM with async-to-LDS staged A tiles (double buffered).
//    Block = 128 threads (4 waves). Block tile 64(M) x 128(N); wave tile 64x32.
//    A tile (64x32 bf16, shared by all 4 waves) is copied with
//    global_load_async_to_lds_b128 (ASYNCcnt) while the current step computes.
//    Rows padded to 40 shorts (80 B) -> conflict-free ds_load_b128 fragments.
// ---------------------------------------------------------------------------
#define APAD 40   // shorts per LDS row (32 data + 8 pad)

__global__ __launch_bounds__(128) void k_gemm_bias_relu(
        const unsigned short* __restrict__ A,   // [M,K] bf16
        const unsigned short* __restrict__ B,   // [N,K] bf16 (einsum 'rd,ed->re')
        const float* __restrict__ bias,         // [N]
        unsigned short* __restrict__ Cout,      // [M,N] bf16
        int M, int N, int K) {
    __shared__ unsigned short sA[2][64 * APAD];

    const int tid  = threadIdx.x;
    const int wave = tid >> 5;
    const int lane = tid & 31;
    const int m0 = blockIdx.y * 64;
    const int n0 = blockIdx.x * 128 + wave * 32;

    const int lrow  = lane & 15;   // M index (A) / N index (B) within tile
    const int lhalf = lane >> 4;   // half-wave selects K sub-chunk

    // async copy mapping: 256 16B chunks; thread handles chunks 2*tid, 2*tid+1
    const int srow  = tid >> 1;          // 0..63
    const int spart = (tid & 1) * 2;     // 0 or 2 (second chunk via offset:16)
    const unsigned lds0 = (unsigned)(uintptr_t)&sA[0][srow * APAD + spart * 8];
    const unsigned lds1 = (unsigned)(uintptr_t)&sA[1][srow * APAD + spart * 8];
    const unsigned short* grow = A + (size_t)(m0 + srow) * K + spart * 8;

    v8f acc[4][2] = {};

    // prologue: stage K-step 0 into buffer 0
    {
        unsigned long long ga = (unsigned long long)(uintptr_t)grow;
        asm volatile("global_load_async_to_lds_b128 %0, %1, off"
                     :: "v"(lds0), "v"(ga) : "memory");
        asm volatile("global_load_async_to_lds_b128 %0, %1, off offset:16"
                     :: "v"(lds0), "v"(ga) : "memory");
    }
    asm volatile("s_wait_asynccnt 0" ::: "memory");
    __syncthreads();

    for (int k0 = 0; k0 < K; k0 += 32) {
        const int cur = (k0 >> 5) & 1;

        // kick off async copy of the next A tile into the other buffer
        if (k0 + 32 < K) {
            unsigned long long ga =
                (unsigned long long)(uintptr_t)(grow + k0 + 32);
            unsigned la = cur ? lds0 : lds1;
            asm volatile("global_load_async_to_lds_b128 %0, %1, off"
                         :: "v"(la), "v"(ga) : "memory");
            asm volatile("global_load_async_to_lds_b128 %0, %1, off offset:16"
                         :: "v"(la), "v"(ga) : "memory");
        }

        v16bf a[4], b[2];
        // B fragment (global): lanes 0-15 hold K {0..7,16..23}, lanes 16-31 {8..15,24..31}
#pragma unroll
        for (int j = 0; j < 2; ++j) {
            const unsigned short* p =
                B + (size_t)(n0 + j * 16 + lrow) * K + k0 + (lhalf << 3);
            AB32 f;
            f.lo = *reinterpret_cast<const uint4*>(p);
            f.hi = *reinterpret_cast<const uint4*>(p + 16);
            b[j] = __builtin_bit_cast(v16bf, f);
        }
        if (k0 + 128 < K)   // global_prefetch_b8 a few K-steps ahead on the B stream
            __builtin_prefetch(B + (size_t)(n0 + lrow) * K + k0 + 128, 0, 1);

        // A fragment from LDS: lane<16 -> K {0..7,16..23}; lane>=16 -> {8..15,24..31}
#pragma unroll
        for (int i = 0; i < 4; ++i) {
            const unsigned short* sp =
                &sA[cur][(i * 16 + lrow) * APAD + (lhalf << 3)];
            AB32 f;
            f.lo = *reinterpret_cast<const uint4*>(sp);
            f.hi = *reinterpret_cast<const uint4*>(sp + 16);
            a[i] = __builtin_bit_cast(v16bf, f);
        }

#pragma unroll
        for (int i = 0; i < 4; ++i)
#pragma unroll
            for (int j = 0; j < 2; ++j)
                acc[i][j] = __builtin_amdgcn_wmma_f32_16x16x32_bf16(
                    false, a[i], false, b[j], (short)0, acc[i][j], false, false);

        // next buffer fully written + all waves past their reads of it
        asm volatile("s_wait_asynccnt 0" ::: "memory");
        __syncthreads();
    }

    // Epilogue: D layout -> vgpr vi, lane L: m = vi + 8*(L>>4), n = L&15
#pragma unroll
    for (int j = 0; j < 2; ++j) {
        int col = n0 + j * 16 + lrow;
        float bv = bias[col];
#pragma unroll
        for (int i = 0; i < 4; ++i) {
#pragma unroll
            for (int vi = 0; vi < 8; ++vi) {
                int row = m0 + i * 16 + vi + (lhalf << 3);
                float v = fmaxf(acc[i][j][vi] + bv, 0.0f);
                Cout[(size_t)row * N + col] = f2bf_bits(v);
            }
        }
    }
}

// ---------------------------------------------------------------------------
// 4) Heads: xc/xd[r,c] = relu( h2[r,:] . w8{c,d}[c,:] + b )  (wave per output)
// ---------------------------------------------------------------------------
__global__ __launch_bounds__(128) void k_heads(const unsigned short* __restrict__ h2b,
                                               const float* __restrict__ w8c,
                                               const float* __restrict__ b8c,
                                               const float* __restrict__ w8d,
                                               const float* __restrict__ b8d,
                                               float* __restrict__ xc,
                                               float* __restrict__ xd) {
    int r    = blockIdx.x;
    int wave = threadIdx.x >> 5;
    int lane = threadIdx.x & 31;
    const unsigned* hp = reinterpret_cast<const unsigned*>(h2b + (size_t)r * DDIM);
    for (int o = wave; o < 2 * NCLS; o += 4) {
        int cls = (o < NCLS) ? o : o - NCLS;
        const float* wp = ((o < NCLS) ? w8c : w8d) + (size_t)cls * DDIM;
        float acc = 0.0f;
        for (int k = lane * 2; k < DDIM; k += 64) {
            unsigned pair = hp[k >> 1];
            float h0 = __builtin_bit_cast(float, pair << 16);
            float h1 = __builtin_bit_cast(float, pair & 0xFFFF0000u);
            acc = fmaf(h0, wp[k], acc);
            acc = fmaf(h1, wp[k + 1], acc);
        }
#pragma unroll
        for (int off = 16; off; off >>= 1) acc += __shfl_xor(acc, off, 32);
        if (lane == 0) {
            float b = (o < NCLS) ? b8c[cls] : b8d[cls];
            float v = fmaxf(acc + b, 0.0f);
            ((o < NCLS) ? xc : xd)[(size_t)r * NCLS + cls] = v;
        }
    }
}

// ---------------------------------------------------------------------------
// 5) Row softmax over 21 classes (wave per ROI): dr = softmax(xc, axis=2)
// ---------------------------------------------------------------------------
__global__ __launch_bounds__(256) void k_row_softmax(const float* __restrict__ xc,
                                                     float* __restrict__ dr) {
    int wave = threadIdx.x >> 5, lane = threadIdx.x & 31;
    int r = blockIdx.x * 8 + wave;
    float v = (lane < NCLS) ? xc[(size_t)r * NCLS + lane] : -1e30f;
    float m = v;
#pragma unroll
    for (int off = 16; off; off >>= 1) m = fmaxf(m, __shfl_xor(m, off, 32));
    float e = (lane < NCLS) ? __expf(v - m) : 0.0f;
    float s = e;
#pragma unroll
    for (int off = 16; off; off >>= 1) s += __shfl_xor(s, off, 32);
    if (lane < NCLS) dr[(size_t)r * NCLS + lane] = e / s;
}

// ---------------------------------------------------------------------------
// 6) Column softmax over 2048 ROIs, dm = dr * softmax(xd, axis=1), score=sum_r
// ---------------------------------------------------------------------------
__global__ __launch_bounds__(256) void k_col_softmax_score(const float* __restrict__ xd,
                                                           const float* __restrict__ dr,
                                                           float* __restrict__ dm,
                                                           float* __restrict__ score) {
    __shared__ float red[256];
    __shared__ float s_m, s_inv;
    int c = blockIdx.x, tid = threadIdx.x;

    float m = -1e30f;
    for (int r = tid; r < R_ROI; r += 256) m = fmaxf(m, xd[(size_t)r * NCLS + c]);
    red[tid] = m; __syncthreads();
    for (int o = 128; o; o >>= 1) { if (tid < o) red[tid] = fmaxf(red[tid], red[tid + o]); __syncthreads(); }
    if (tid == 0) s_m = red[0];
    __syncthreads();
    m = s_m;

    float s = 0.0f;
    for (int r = tid; r < R_ROI; r += 256) s += __expf(xd[(size_t)r * NCLS + c] - m);
    red[tid] = s; __syncthreads();
    for (int o = 128; o; o >>= 1) { if (tid < o) red[tid] += red[tid + o]; __syncthreads(); }
    if (tid == 0) s_inv = 1.0f / red[0];
    __syncthreads();
    float inv = s_inv;

    float sc = 0.0f;
    for (int r = tid; r < R_ROI; r += 256) {
        size_t idx = (size_t)r * NCLS + c;
        float d = dr[idx] * (__expf(xd[idx] - m) * inv);
        dm[idx] = d;
        sc += d;
    }
    red[tid] = sc; __syncthreads();
    for (int o = 128; o; o >>= 1) { if (tid < o) red[tid] += red[tid + o]; __syncthreads(); }
    if (tid == 0) score[c] = red[0];
}

// ---------------------------------------------------------------------------
// launch
// ---------------------------------------------------------------------------
extern "C" void kernel_launch(void* const* d_in, const int* in_sizes, int n_in,
                              void* d_out, int out_size, void* d_ws, size_t ws_size,
                              hipStream_t stream) {
    (void)in_sizes; (void)n_in; (void)out_size; (void)ws_size;

    const float* x   = (const float*)d_in[0];
    const float* w6  = (const float*)d_in[1];
    const float* b6  = (const float*)d_in[2];
    const float* w7  = (const float*)d_in[3];
    const float* b7  = (const float*)d_in[4];
    const float* w8c = (const float*)d_in[5];
    const float* b8c = (const float*)d_in[6];
    const float* w8d = (const float*)d_in[7];
    const float* b8d = (const float*)d_in[8];
    const int*   ssw = (const int*)d_in[9];

    float* out = (float*)d_out;
    float* dm    = out;                       // [2048*21]
    float* dr    = out + R_ROI * NCLS;        // [2048*21]
    float* score = out + 2 * R_ROI * NCLS;    // [21]

    // workspace carve-up (bf16 weights/activations + fp32 head outputs)
    char* ws = (char*)d_ws;
    unsigned short* w6b = (unsigned short*)(ws);                          // 32 MB
    unsigned short* w7b = (unsigned short*)(ws + (size_t)33554432);       // 32 MB
    unsigned short* yb  = (unsigned short*)(ws + (size_t)67108864);       // 16 MB
    unsigned short* h1b = (unsigned short*)(ws + (size_t)83886080);       // 16 MB
    unsigned short* h2b = (unsigned short*)(ws + (size_t)100663296);      // 16 MB
    float*          xc  = (float*)(ws + (size_t)117440512);               // 168 KB
    float*          xd  = (float*)(ws + (size_t)117612544);               // 168 KB

    // 1) weights -> bf16 (64 MB HBM reads each; after this everything lives in L2)
    {
        int n8 = (DDIM * DDIM) / 8;
        int blk = 256, grd = (n8 + blk - 1) / blk;
        k_f32_to_bf16<<<grd, blk, 0, stream>>>(w6, w6b, n8);
        k_f32_to_bf16<<<grd, blk, 0, stream>>>(w7, w7b, n8);
    }
    // 2) SPP -> bf16 features
    k_spp<<<dim3(R_ROI, C_CH / 64), 256, 0, stream>>>(x, ssw, yb);

    // 3) FC6 / FC7 (the 137-GFLOP core: bf16 WMMA, async-LDS A tiles)
    dim3 ggrid(DDIM / 128, R_ROI / 64);
    k_gemm_bias_relu<<<ggrid, 128, 0, stream>>>(yb,  w6b, b6, h1b, R_ROI, DDIM, DDIM);
    k_gemm_bias_relu<<<ggrid, 128, 0, stream>>>(h1b, w7b, b7, h2b, R_ROI, DDIM, DDIM);

    // 4) classifier + detector heads
    k_heads<<<R_ROI, 128, 0, stream>>>(h2b, w8c, b8c, w8d, b8d, xc, xd);

    // 5) softmax over classes -> dr
    k_row_softmax<<<R_ROI / 8, 256, 0, stream>>>(xc, dr);

    // 6) softmax over ROIs, dm = dr * sm(xd), score = sum_r dm
    k_col_softmax_score<<<NCLS, 256, 0, stream>>>(xd, dr, dm, score);
}
